// SelfAttention_69277822484981
// MI455X (gfx1250) — matrix operand order repaired and checked by reading, not compiled, not run on hardware
//
#include <hip/hip_runtime.h>
#include <hip/hip_bf16.h>
#include <stdint.h>

// ---------------------------------------------------------------------------
// Problem constants (fixed by the reference): B=4, S=2048, D=K=1024
// ---------------------------------------------------------------------------
#define BATCH 4
#define SEQ   2048
#define DIM   1024
#define ROWS  (BATCH * SEQ)     // 8192

typedef __attribute__((ext_vector_type(16))) _Float16 v16h;
typedef __attribute__((ext_vector_type(8)))  _Float16 v8h;
typedef __attribute__((ext_vector_type(4)))  _Float16 v4h;
typedef __attribute__((ext_vector_type(8)))  float    v8f;
typedef int v4i __attribute__((vector_size(16)));

// ---------------------------------------------------------------------------
// CDNA5 async global->LDS copy (ASYNCcnt-tracked), with sync fallback.
// Probe-confirmed signature: (int4 __device__* src, int4 __shared__* dst,
//                             imm int offset, imm int cpol)
// ---------------------------------------------------------------------------
#if __has_builtin(__builtin_amdgcn_global_load_async_to_lds_b128)
#define HAVE_ASYNC_LDS 1
#else
#define HAVE_ASYNC_LDS 0
#endif

#if HAVE_ASYNC_LDS
#define ASYNC_CP16(ldsp, gp)                                                   \
    __builtin_amdgcn_global_load_async_to_lds_b128(                            \
        (__attribute__((address_space(1))) v4i*)(gp),                          \
        (__attribute__((address_space(3))) v4i*)(ldsp), 0, 0)
#if __has_builtin(__builtin_amdgcn_s_wait_asynccnt)
#define WAIT_ASYNC(n) __builtin_amdgcn_s_wait_asynccnt(n)
#else
#define WAIT_ASYNC(n) asm volatile("s_wait_asynccnt %0" ::"i"(n))
#endif
#endif

// ---------------------------------------------------------------------------
// WMMA helpers (CDNA5 16x16x32 f16 -> f32)
// A frag (ISA 7.12.2): lane m=lane&15; halves 0..7 = K 8g..8g+7, halves 8..15 =
//   K 16+8g..16+8g+7 (g=lane>>4)  => two 16B loads at p, p+16
// B frag: lane n=lane&15; same K order as A frag => two 16B loads at p, p+16
// C/D frag: VGPR r holds row M = r + 8g, col N = lane&15
// ---------------------------------------------------------------------------
__device__ __forceinline__ v16h load_a_frag(const _Float16* p) {
    union { v16h v; v8h h[2]; } u;
    u.h[0] = *(const v8h*)(p);
    u.h[1] = *(const v8h*)(p + 16);
    return u.v;
}

__device__ __forceinline__ v16h load_b_frag(const _Float16* p) {
    union { v16h v; v8h h[2]; } u;
    u.h[0] = *(const v8h*)(p);
    u.h[1] = *(const v8h*)(p + 16);
    return u.v;
}

__device__ __forceinline__ v8f wmma_f16(v16h a, v16h b, v8f c) {
    return __builtin_amdgcn_wmma_f32_16x16x32_f16(
        false, a, false, b, (short)0, c, false, false);
}

__device__ __forceinline__ v8f v8f_zero() {
    v8f z = {0.f, 0.f, 0.f, 0.f, 0.f, 0.f, 0.f, 0.f};
    return z;
}

// ---------------------------------------------------------------------------
// Kernel 0: positional encoding + LayerNorm -> f16 activations
// one block (256 threads) per row of [B*S, D]; 4 consecutive elems per thread
// ---------------------------------------------------------------------------
__global__ __launch_bounds__(256) void posln_kernel(
    const float* __restrict__ x, const float* __restrict__ lnw,
    const float* __restrict__ lnb, _Float16* __restrict__ xn)
{
    const int row = blockIdx.x;
    const int s   = row & (SEQ - 1);
    const int tid = threadIdx.x;
    const int d0  = tid * 4;
    const float4 xv = *(const float4*)(x + (size_t)row * DIM + d0);

    float h[4];
    float sum = 0.f, sq = 0.f;
#pragma unroll
    for (int i = 0; i < 4; ++i) {
        const int d = d0 + i;
        const float ex   = (2.0f * (float)d) * (1.0f / (float)DIM);
        const float freq = __expf(-ex * 9.210340371976184f);   // ln(10000)
        const float ang  = (float)s * freq;
        const float pe   = (d & 1) ? __cosf(ang) : __sinf(ang);
        const float v = pe + ((const float*)&xv)[i];
        h[i] = v; sum += v; sq += v * v;
    }
#pragma unroll
    for (int m = 1; m < 32; m <<= 1) {
        sum += __shfl_xor(sum, m, 32);
        sq  += __shfl_xor(sq,  m, 32);
    }
    __shared__ float s1[8], s2[8];
    const int w = tid >> 5, lane = tid & 31;
    if (lane == 0) { s1[w] = sum; s2[w] = sq; }
    __syncthreads();
    if (w == 0) {
        float a  = (lane < 8) ? s1[lane] : 0.f;
        float b2 = (lane < 8) ? s2[lane] : 0.f;
#pragma unroll
        for (int m = 1; m < 8; m <<= 1) {
            a  += __shfl_xor(a,  m, 32);
            b2 += __shfl_xor(b2, m, 32);
        }
        if (lane == 0) { s1[0] = a; s2[0] = b2; }
    }
    __syncthreads();
    const float mean = s1[0] * (1.0f / (float)DIM);
    const float var  = s2[0] * (1.0f / (float)DIM) - mean * mean;
    const float rstd = rsqrtf(var + 1e-5f);

    const float4 wv = *(const float4*)(lnw + d0);
    const float4 bv = *(const float4*)(lnb + d0);
    v4h ov;
#pragma unroll
    for (int i = 0; i < 4; ++i)
        ov[i] = (_Float16)(((h[i] - mean) * rstd) * ((const float*)&wv)[i] +
                           ((const float*)&bv)[i]);
    *(v4h*)(xn + (size_t)row * DIM + d0) = ov;
}

// ---------------------------------------------------------------------------
// Kernel 0b: Wk [k][n] f32 -> wkT [n][k] f16 (transpose + convert), 32x32 tiles
// grid: (D/32, D/32)
// ---------------------------------------------------------------------------
__global__ __launch_bounds__(256) void wk_transpose_kernel(
    const float* __restrict__ Wk, _Float16* __restrict__ wkT)
{
    __shared__ __align__(16) _Float16 t[32][36];
    const int k0 = blockIdx.x * 32;
    const int n0 = blockIdx.y * 32;
    const int tid = threadIdx.x;
    const int r  = tid >> 3;          // 0..31
    const int c4 = (tid & 7) * 4;     // 0..28

    const float4 f = *(const float4*)(Wk + (size_t)(k0 + r) * DIM + n0 + c4);
    v4h p;
    p[0] = (_Float16)f.x; p[1] = (_Float16)f.y;
    p[2] = (_Float16)f.z; p[3] = (_Float16)f.w;
    *(v4h*)&t[r][c4] = p;
    __syncthreads();
    v4h o;
#pragma unroll
    for (int i = 0; i < 4; ++i) o[i] = t[c4 + i][r];
    *(v4h*)(wkT + (size_t)(n0 + r) * DIM + k0 + c4) = o;
}

// ---------------------------------------------------------------------------
// Kernel 1: qkv = xn @ wkT^T + bk  (M=8192, N=1024, K=1024), output f16
// block tile 128x128, 8 waves each 32x64 (2x4 WMMA frags), K step 32
// double-buffered LDS staged with GLOBAL_LOAD_ASYNC_TO_LDS_B128
// grid: (N/128, M/128) = (8, 64)
// ---------------------------------------------------------------------------
__global__ __launch_bounds__(256) void gemm_qkv_kernel(
    const _Float16* __restrict__ xn, const _Float16* __restrict__ wkT,
    const float* __restrict__ bk, _Float16* __restrict__ qkv)
{
    __shared__ __align__(16) _Float16 Ash[2][128][40];   // [m][k]
    __shared__ __align__(16) _Float16 Bsh[2][128][40];   // [n][k]

    const int m0  = blockIdx.y * 128;
    const int n0  = blockIdx.x * 128;
    const int tid = threadIdx.x;
    const int w   = tid >> 5, lane = tid & 31;
    const int wm  = w >> 1,   wn   = w & 1;
    const int gc  = lane >> 4, ln16 = lane & 15;

    // per-thread staging assignment: 512 chunks of 16B per tile, 2 per thread
    const int cA = tid,       cB = tid + 256;
    const int r0 = cA >> 2,   o0 = (cA & 3) * 8;
    const int r1 = cB >> 2,   o1 = (cB & 3) * 8;
    const _Float16* gA0 = xn  + (size_t)(m0 + r0) * DIM + o0;
    const _Float16* gA1 = xn  + (size_t)(m0 + r1) * DIM + o1;
    const _Float16* gB0 = wkT + (size_t)(n0 + r0) * DIM + o0;
    const _Float16* gB1 = wkT + (size_t)(n0 + r1) * DIM + o1;

    v8f acc[2][4];
#pragma unroll
    for (int i = 0; i < 2; ++i)
#pragma unroll
        for (int j = 0; j < 4; ++j) acc[i][j] = v8f_zero();

#if HAVE_ASYNC_LDS
#define STAGE(buf, kk)                                                         \
    do {                                                                       \
        ASYNC_CP16(&Ash[buf][r0][o0], gA0 + (kk));                             \
        ASYNC_CP16(&Ash[buf][r1][o1], gA1 + (kk));                             \
        ASYNC_CP16(&Bsh[buf][r0][o0], gB0 + (kk));                             \
        ASYNC_CP16(&Bsh[buf][r1][o1], gB1 + (kk));                             \
    } while (0)
#endif
#define STAGE_SYNC(buf, kk)                                                    \
    do {                                                                       \
        *(v8h*)&Ash[buf][r0][o0] = *(const v8h*)(gA0 + (kk));                  \
        *(v8h*)&Ash[buf][r1][o1] = *(const v8h*)(gA1 + (kk));                  \
        *(v8h*)&Bsh[buf][r0][o0] = *(const v8h*)(gB0 + (kk));                  \
        *(v8h*)&Bsh[buf][r1][o1] = *(const v8h*)(gB1 + (kk));                  \
    } while (0)

#if HAVE_ASYNC_LDS
    STAGE(0, 0);
#endif
    for (int kk = 0; kk < DIM; kk += 32) {
        const int cur = (kk >> 5) & 1;
        __syncthreads();   // all waves finished reading buffer 'cur^1'
#if HAVE_ASYNC_LDS
        if (kk + 32 < DIM) {
            STAGE(cur ^ 1, kk + 32);   // prefetch next tile (stays in flight)
            WAIT_ASYNC(4);             // this tile's 4 copies complete
        } else {
            WAIT_ASYNC(0);
        }
#else
        STAGE_SYNC(cur, kk);
#endif
        __syncthreads();   // every wave's copies for 'cur' are visible

        v16h a[2], b[4];
#pragma unroll
        for (int fm = 0; fm < 2; ++fm)
            a[fm] = load_a_frag(&Ash[cur][wm * 32 + fm * 16 + ln16][8 * gc]);
#pragma unroll
        for (int fn = 0; fn < 4; ++fn)
            b[fn] = load_b_frag(&Bsh[cur][wn * 64 + fn * 16 + ln16][8 * gc]);
#pragma unroll
        for (int fm = 0; fm < 2; ++fm)
#pragma unroll
            for (int fn = 0; fn < 4; ++fn)
                acc[fm][fn] = wmma_f16(a[fm], b[fn], acc[fm][fn]);
    }

#pragma unroll
    for (int fm = 0; fm < 2; ++fm)
#pragma unroll
        for (int fn = 0; fn < 4; ++fn) {
            const int nn   = n0 + wn * 64 + fn * 16 + ln16;
            const float bv = bk[nn];
#pragma unroll
            for (int r = 0; r < 8; ++r) {
                const int mm = m0 + wm * 32 + fm * 16 + r + 8 * gc;
                qkv[(size_t)mm * DIM + nn] = (_Float16)(acc[fm][fn][r] + bv);
            }
        }
}

// ---------------------------------------------------------------------------
// Kernel 1b: transpose qkv [B][S][D] -> qkvT [B][D][S] (f16), 32x32 tiles
// grid: (S/32, D/32, B)
// ---------------------------------------------------------------------------
__global__ __launch_bounds__(256) void transpose_kernel(
    const _Float16* __restrict__ qkv, _Float16* __restrict__ qkvT)
{
    __shared__ __align__(16) _Float16 t[32][36];
    const int b  = blockIdx.z;
    const int s0 = blockIdx.x * 32;
    const int d0 = blockIdx.y * 32;
    const int tid = threadIdx.x;
    const int r  = tid >> 3;
    const int c4 = (tid & 7) * 4;

    *(v4h*)&t[r][c4] =
        *(const v4h*)(qkv + ((size_t)b * SEQ + s0 + r) * DIM + d0 + c4);
    __syncthreads();
    v4h o;
#pragma unroll
    for (int i = 0; i < 4; ++i) o[i] = t[c4 + i][r];
    *(v4h*)(qkvT + ((size_t)b * DIM + d0 + r) * SEQ + s0 + c4) = o;
}

// ---------------------------------------------------------------------------
// Kernel 2: flash attention.  scores = (Q K^T)/32, masked->1e-9, softmax, @V
// block = 32 query rows; 8 waves = 2 row-groups(16) x 4 col-blocks(256)
// QK^T split-K across the 4 vc waves, reduced through LDS.
// grid: B * S/32 = 256 blocks
// ---------------------------------------------------------------------------
__global__ __launch_bounds__(256) void attn_kernel(
    const _Float16* __restrict__ qkv, const _Float16* __restrict__ qkvT,
    const unsigned char* __restrict__ mask, float* __restrict__ out)
{
    __shared__ __align__(16) float    sbuf[2][2][8][32][4]; // [qg][f][r][lane][vc]
    __shared__ __align__(16) _Float16 pbuf[2][16][48];      // per-qg P tile

    const int blk = blockIdx.x;
    const int b   = blk >> 6;
    const int q0  = (blk & 63) * 32;
    const int tid = threadIdx.x;
    const int w   = tid >> 5, lane = tid & 31;
    const int qg  = w >> 2,  vc   = w & 3;
    const int gc  = lane >> 4, ln16 = lane & 15;

    const _Float16* Q  = qkv  + ((size_t)b * SEQ + q0 + qg * 16) * DIM;
    const _Float16* KV = qkv  + (size_t)b * SEQ * DIM;
    const _Float16* VT = qkvT + (size_t)b * DIM * SEQ;
    const unsigned char* mb = mask + (size_t)b * SEQ * SEQ;

    v8f o[16];
#pragma unroll
    for (int i = 0; i < 16; ++i) o[i] = v8f_zero();
    float rmax[8], rsum[8];
#pragma unroll
    for (int r = 0; r < 8; ++r) { rmax[r] = -3.0e38f; rsum[r] = 0.f; }

    for (int kt0 = 0; kt0 < SEQ; kt0 += 32) {
        // ---- partial scores over this wave's K-range (256 of 1024) ----
        v8f sc[2];
        sc[0] = v8f_zero(); sc[1] = v8f_zero();
        const int c0 = vc * 256;
#pragma unroll
        for (int ci = 0; ci < 8; ++ci) {
            const int cc = c0 + ci * 32;
            const v16h a = load_a_frag(Q + (size_t)ln16 * DIM + cc + 8 * gc);
#pragma unroll
            for (int f = 0; f < 2; ++f) {
                const v16h bf = load_b_frag(
                    KV + (size_t)(kt0 + f * 16 + ln16) * DIM + cc + 8 * gc);
                sc[f] = wmma_f16(a, bf, sc[f]);
            }
        }
        // ---- publish partials, reduce across vc through LDS ----
#pragma unroll
        for (int f = 0; f < 2; ++f)
#pragma unroll
            for (int r = 0; r < 8; ++r)
                sbuf[qg][f][r][lane][vc] = sc[f][r];
        __syncthreads();                               // (A)
#pragma unroll
        for (int f = 0; f < 2; ++f)
#pragma unroll
            for (int r = 0; r < 8; ++r) {
                const float4 t = *(const float4*)&sbuf[qg][f][r][lane][0];
                sc[f][r] = t.x + t.y + t.z + t.w;
            }
        // ---- scale by 1/sqrt(1024), masked_fill(1e-9) ----
#pragma unroll
        for (int f = 0; f < 2; ++f)
#pragma unroll
            for (int r = 0; r < 8; ++r) {
                const int qi = q0 + qg * 16 + r + 8 * gc;
                const int si = kt0 + f * 16 + ln16;
                float v = sc[f][r] * 0.03125f;
                if (mb[(size_t)qi * SEQ + si]) v = 1e-9f;
                sc[f][r] = v;
            }
        // ---- online softmax (rows r+8*gc live across the 16-lane half) ----
        float tm[8];
#pragma unroll
        for (int r = 0; r < 8; ++r) tm[r] = fmaxf(sc[0][r], sc[1][r]);
#pragma unroll
        for (int m = 1; m < 16; m <<= 1)
#pragma unroll
            for (int r = 0; r < 8; ++r)
                tm[r] = fmaxf(tm[r], __shfl_xor(tm[r], m, 32));
        float alpha[8];
#pragma unroll
        for (int r = 0; r < 8; ++r) {
            const float mn = fmaxf(rmax[r], tm[r]);
            alpha[r] = __expf(rmax[r] - mn);
            rmax[r] = mn;
            sc[0][r] = __expf(sc[0][r] - mn);
            sc[1][r] = __expf(sc[1][r] - mn);
        }
        float ps[8];
#pragma unroll
        for (int r = 0; r < 8; ++r) ps[r] = sc[0][r] + sc[1][r];
#pragma unroll
        for (int m = 1; m < 16; m <<= 1)
#pragma unroll
            for (int r = 0; r < 8; ++r) ps[r] += __shfl_xor(ps[r], m, 32);
#pragma unroll
        for (int r = 0; r < 8; ++r) rsum[r] = rsum[r] * alpha[r] + ps[r];
#pragma unroll
        for (int fn = 0; fn < 16; ++fn)
#pragma unroll
            for (int r = 0; r < 8; ++r) o[fn][r] *= alpha[r];

        // ---- P -> f16 in LDS (one wave per qg), C-layout -> A-layout ----
        if (vc == 0) {
#pragma unroll
            for (int f = 0; f < 2; ++f)
#pragma unroll
                for (int r = 0; r < 8; ++r)
                    pbuf[qg][r + 8 * gc][f * 16 + ln16] = (_Float16)sc[f][r];
        }
        __syncthreads();                               // (B)
        const v16h pa = load_a_frag(&pbuf[qg][ln16][8 * gc]);

        // ---- O += P V  (V^T is K-contiguous: vector B-frag loads) ----
#pragma unroll
        for (int fn = 0; fn < 16; ++fn) {
            const int col = vc * 256 + fn * 16 + ln16;
            const v16h bf = load_b_frag(VT + (size_t)col * SEQ + kt0 + 8 * gc);
            o[fn] = wmma_f16(pa, bf, o[fn]);
        }
    }

    // ---- normalize and store f32 output ----
#pragma unroll
    for (int fn = 0; fn < 16; ++fn)
#pragma unroll
        for (int r = 0; r < 8; ++r) {
            const int qi  = q0 + qg * 16 + r + 8 * gc;
            const int col = vc * 256 + fn * 16 + ln16;
            out[((size_t)b * SEQ + qi) * DIM + col] = o[fn][r] / rsum[r];
        }
}

// ---------------------------------------------------------------------------
// Host launcher
// ---------------------------------------------------------------------------
extern "C" void kernel_launch(void* const* d_in, const int* in_sizes, int n_in,
                              void* d_out, int out_size, void* d_ws, size_t ws_size,
                              hipStream_t stream)
{
    const float*         x    = (const float*)d_in[0];          // [4,2048,1024]
    const unsigned char* mask = (const unsigned char*)d_in[1];  // [4,2048,2048] bool
    const float*         lnw  = (const float*)d_in[2];          // [1024]
    const float*         lnb  = (const float*)d_in[3];          // [1024]
    const float*         Wk   = (const float*)d_in[4];          // [1024,1024]
    const float*         bk   = (const float*)d_in[5];          // [1024]
    float*               out  = (float*)d_out;                  // [4,2048,1024] f32

    const size_t half_bytes = (size_t)ROWS * DIM * 2;           // 16 MB
    _Float16* xn   = (_Float16*)d_ws;
    _Float16* qkv  = (_Float16*)((char*)d_ws + half_bytes);
    _Float16* qkvT = (_Float16*)((char*)d_ws + 2 * half_bytes);
    _Float16* wkT  = (_Float16*)((char*)d_ws + 3 * half_bytes); // 2 MB

    posln_kernel<<<ROWS, 256, 0, stream>>>(x, lnw, lnb, xn);

    dim3 gw(DIM / 32, DIM / 32);      // (32, 32)
    wk_transpose_kernel<<<gw, 256, 0, stream>>>(Wk, wkT);

    dim3 g1(DIM / 128, ROWS / 128);   // (8, 64)
    gemm_qkv_kernel<<<g1, 256, 0, stream>>>(xn, wkT, bk, qkv);

    dim3 gt(SEQ / 32, DIM / 32, BATCH);   // (64, 32, 4)
    transpose_kernel<<<gt, 256, 0, stream>>>(qkv, qkvT);

    attn_kernel<<<BATCH * (SEQ / 32), 256, 0, stream>>>(qkv, qkvT, mask, out);
}
